// TransformerBlock_45363444580494
// MI455X (gfx1250) — matrix operand order repaired and checked by reading, not compiled, not run on hardware
//
#include <hip/hip_runtime.h>
#include <math.h>

// ---------------------------------------------------------------------------
// Transformer block (attention + top-2 MoE) for gfx1250 (MI455X).
// All GEMMs use V_WMMA_F32_16X16X32_BF16 (bf16 A/B, fp32 accumulate).
// fp32 weights are converted to bf16 on the fly (read-once, memory-bound).
// MoE is routed (grouped GEMM over gathered token rows, 16-row padded per
// expert so WMMA tiles never span experts).
// ---------------------------------------------------------------------------

#define BB 4
#define TT 1024
#define DD 1024
#define HH 16
#define HD 64
#define EE 8
#define KTOP 2
#define FF 4096
#define BT (BB*TT)           // 4096 rows
#define N3 (3*DD)            // 3072
#define EPSV 1e-5f
#define PAIR_CAP 8320        // 2*BT + E*15 padding, rounded to 16
#define PAIR_TILES (PAIR_CAP/16)
#define NEG_BIG 3.0e38f

typedef __bf16 bf16_t;
typedef __attribute__((ext_vector_type(8)))  __bf16 v8bf;
typedef __attribute__((ext_vector_type(16))) __bf16 v16bf;
typedef __attribute__((ext_vector_type(8)))  float  v8f;

// ---------------- workspace layout (bytes) ---------------------------------
// Live ranges allow aliasing abuf over [h | qkv | o | q] (all dead before MoE).
static constexpr size_t O_H     = 0;                       // bf16 [BT,D]      8.39 MB
static constexpr size_t O_QKV   = O_H   + (size_t)BT*DD*2; // f32  [BT,3D]    50.33 MB
static constexpr size_t O_O     = O_QKV + (size_t)BT*N3*4; // bf16 [BT,D]      8.39 MB
static constexpr size_t O_Q     = O_O   + (size_t)BT*DD*2; // bf16 [B,H,T,HD]  8.39 MB
static constexpr size_t O_K     = O_Q   + (size_t)BT*DD*2;
static constexpr size_t O_V     = O_K   + (size_t)BT*DD*2;
static constexpr size_t O_X2    = O_V   + (size_t)BT*DD*2; // f32  [BT,D]     16.78 MB
static constexpr size_t O_H2    = O_X2  + (size_t)BT*DD*4; // bf16 [BT,D]
static constexpr size_t O_ROUTE = O_H2  + (size_t)BT*DD*2; // 256 KB routing
static constexpr size_t O_MOE   = O_ROUTE + 262144;        // f32  [BT,D]
static constexpr size_t O_ABUF  = 0;  // bf16 [PAIR_CAP,FF] = 68.2MB, aliases h..q
// routing sub-offsets (relative to O_ROUTE)
static constexpr size_t R_COUNTS = 0;
static constexpr size_t R_OFFS   = 64;
static constexpr size_t R_FILL   = 128;
static constexpr size_t R_TOKE   = 256;                       // int [BT*2]
static constexpr size_t R_TOKP   = R_TOKE + (size_t)BT*2*4;   // f32 [BT*2]
static constexpr size_t R_PTOK   = R_TOKP + (size_t)BT*2*4;   // int [PAIR_CAP]
static constexpr size_t R_PWT    = R_PTOK + (size_t)PAIR_CAP*4;
static constexpr size_t R_REXP   = R_PWT  + (size_t)PAIR_CAP*4;

// ---------------- WMMA fragment helpers ------------------------------------
__device__ inline v8f wmma_bf16(v16bf a, v16bf b, v8f c) {
  return __builtin_amdgcn_wmma_f32_16x16x32_bf16(false, a, false, b, (short)0, c,
                                                 false, false);
}
__device__ inline v16bf cat8(v8bf lo, v8bf hi) {
  v16bf r;
#pragma unroll
  for (int i = 0; i < 8; ++i) { r[i] = lo[i]; r[i + 8] = hi[i]; }
  return r;
}
// A-matrix 16x32 bf16: lane holds row (lane&15); elem e -> K=kb+16g+e
// => two contiguous 16-byte loads.
__device__ inline v16bf afrag(const bf16_t* rowp, int kb, int grp) {
  v8bf lo = *(const v8bf*)(rowp + kb + 16 * grp);
  v8bf hi = *(const v8bf*)(rowp + kb + 16 * grp + 8);
  return cat8(lo, hi);
}
// B-matrix 32x16 bf16: lane holds col (lane&15); elem e -> K = 16*grp + e.
// Contiguous when source is [n][k]-major:
__device__ inline v16bf bfrag_contig(const bf16_t* p) {
  return cat8(*(const v8bf*)p, *(const v8bf*)(p + 8));
}
// Gathered B-frag from fp32 [k][n] weights (converted on the fly). Lanes 0..15
// read 64 consecutive bytes per element step -> coalesced 2x64B segments.
__device__ inline v16bf bfrag_strided_f32(const float* p, int ld) {
  v16bf r;
#pragma unroll
  for (int e = 0; e < 16; ++e) r[e] = (__bf16)p[(size_t)e * ld];
  return r;
}
__device__ inline void atomic_add_f32(float* p, float v) {
  __hip_atomic_fetch_add(p, v, __ATOMIC_RELAXED, __HIP_MEMORY_SCOPE_AGENT);
}

// ---------------- init ------------------------------------------------------
__global__ void init_kernel(float* moe, int* pair_tok, int* fill, int* counts) {
  size_t i = (size_t)blockIdx.x * blockDim.x + threadIdx.x;
  size_t stride = (size_t)gridDim.x * blockDim.x;
  for (size_t j = i; j < (size_t)BT * DD; j += stride) moe[j] = 0.0f;
  if (i < PAIR_CAP) pair_tok[i] = -1;
  if (i < EE) { fill[i] = 0; counts[i] = 0; }
}

// ---------------- rmsnorm (row-per-block) -----------------------------------
__global__ void rmsnorm_kernel(const float* __restrict__ x,
                               const float* __restrict__ scale,
                               bf16_t* __restrict__ y) {
  __shared__ float red[256];
  const int row = blockIdx.x;
  const float* xr = x + (size_t)row * DD;
  float s = 0.0f;
  for (int d = threadIdx.x; d < DD; d += 256) { float v = xr[d]; s += v * v; }
  red[threadIdx.x] = s;
  __syncthreads();
  for (int st = 128; st > 0; st >>= 1) {
    if (threadIdx.x < st) red[threadIdx.x] += red[threadIdx.x + st];
    __syncthreads();
  }
  const float norm = sqrtf(red[0]) * (1.0f / 32.0f);  // /sqrt(D)
  const float inv = 1.0f / (norm + EPSV);
  for (int d = threadIdx.x; d < DD; d += 256)
    y[(size_t)row * DD + d] = (__bf16)(xr[d] * inv * scale[d]);
}

// ---------------- generic GEMM: C = A(bf16) x B(f32->bf16) + bias [+res] ----
// block = 256 thr (8 waves), block tile 64x32, wave tile 16x16, K-chunk 32.
#define BT_STRIDE 40  // 80B: 16B-aligned, conflict-free b128 LDS reads
__global__ void gemm_bf16(const bf16_t* __restrict__ A,
                          const float* __restrict__ Bw,
                          const float* __restrict__ bias,
                          const float* __restrict__ residual,
                          float* __restrict__ C, int Ncols, int Kdim) {
  __shared__ bf16_t Bt[32 * BT_STRIDE];
  const int tid = threadIdx.x;
  const int lane = tid & 31, wid = tid >> 5;
  const int grp = lane >> 4, ln = lane & 15;
  const int mw = wid & 3, nw = wid >> 2;
  const int m0 = blockIdx.y * 64 + mw * 16;
  const int nblk = blockIdx.x * 32;
  const int n0 = nblk + nw * 16;
  const bf16_t* Arow = A + (size_t)(m0 + ln) * Kdim;
  v8f acc = {};
  for (int k0 = 0; k0 < Kdim; k0 += 32) {
    __syncthreads();
#pragma unroll
    for (int j = 0; j < 4; ++j) {          // stage B chunk [32k][32n] -> Bt[n][k]
      int i = tid + j * 256;
      int kk = i >> 5, nn = i & 31;
      Bt[nn * BT_STRIDE + kk] = (__bf16)Bw[(size_t)(k0 + kk) * Ncols + nblk + nn];
    }
    __syncthreads();
    if (k0 + 32 < Kdim)
      __builtin_prefetch(&Bw[(size_t)(k0 + 32) * Ncols + nblk], 0, 1);
    v16bf a = afrag(Arow, k0, grp);
    v16bf b = bfrag_contig(&Bt[(nw * 16 + ln) * BT_STRIDE + grp * 16]);
    acc = wmma_bf16(a, b, acc);
  }
  const int col = n0 + ln;
#pragma unroll
  for (int r = 0; r < 8; ++r) {
    int row = m0 + r + 8 * grp;
    float v = acc[r] + bias[col];
    if (residual) v += residual[(size_t)row * Ncols + col];
    C[(size_t)row * Ncols + col] = v;
  }
}

// ---------------- RoPE + split to q/k/v [B,H,T,HD] bf16 ---------------------
__global__ void rope_split(const float* __restrict__ qkv, bf16_t* __restrict__ q,
                           bf16_t* __restrict__ k, bf16_t* __restrict__ v) {
  int id = blockIdx.x * blockDim.x + threadIdx.x;  // B*T*H*32 threads
  if (id >= BT * HH * 32) return;
  const int i = id & 31;
  const int h = (id >> 5) & (HH - 1);
  const int n = id >> 9;          // b*T + t
  const int b = n >> 10, t = n & (TT - 1);
  const float theta = powf(10000.0f, -(2.0f * i) / (float)HD);
  float s, c;
  sincosf((float)t * theta, &s, &c);
  const float* src = qkv + (size_t)n * N3 + h * HD;
  const size_t dst = ((size_t)(b * HH + h) * TT + t) * HD;
  float q1 = src[0 * DD + i], q2 = src[0 * DD + i + 32];
  q[dst + i] = (__bf16)(q1 * c - q2 * s);
  q[dst + i + 32] = (__bf16)(q1 * s + q2 * c);
  float k1 = src[1 * DD + i], k2 = src[1 * DD + i + 32];
  k[dst + i] = (__bf16)(k1 * c - k2 * s);
  k[dst + i + 32] = (__bf16)(k1 * s + k2 * c);
  v[dst + i] = (__bf16)src[2 * DD + i];
  v[dst + i + 32] = (__bf16)src[2 * DD + i + 32];
}

// ---------------- flash attention, 1 wave per 16-row Q tile ------------------
// S = Q·K^T via WMMA (2 d-chunks), online softmax with 16-lane butterflies,
// P bounced through LDS (C-layout -> A-layout), O += P·V (V staged in LDS).
__global__ void __launch_bounds__(32) attn_kernel(const bf16_t* __restrict__ qb,
                                                  const bf16_t* __restrict__ kb,
                                                  const bf16_t* __restrict__ vb,
                                                  bf16_t* __restrict__ ob) {
  __shared__ bf16_t P[16 * 32];
  __shared__ bf16_t Vt[32 * 72];  // 144B row stride (16B aligned)
  const int lane = threadIdx.x & 31;
  const int grp = lane >> 4, ln = lane & 15;
  const int qt = blockIdx.x, h = blockIdx.y, b = blockIdx.z;
  const size_t bh = (size_t)(b * HH + h);
  const bf16_t* qrowp = qb + (bh * TT + qt * 16 + ln) * HD;
  const v16bf qa0 = afrag(qrowp, 0, grp);
  const v16bf qa1 = afrag(qrowp, 32, grp);
  v8f o0 = {}, o1 = {}, o2 = {}, o3 = {};
  float mrow[8], lrow[8];
#pragma unroll
  for (int r = 0; r < 8; ++r) { mrow[r] = -NEG_BIG; lrow[r] = 0.0f; }
  const int kmax = qt * 16 + 15;
  for (int k0 = 0; k0 <= kmax; k0 += 32) {
    v8f s0 = {}, s1 = {};
    {  // Kᵀ B-frags: col = key, K-dim = head dim (contiguous in memory)
      const int key0 = min(k0 + ln, TT - 1);
      const int key1 = min(k0 + 16 + ln, TT - 1);
      const bf16_t* kp0 = kb + (bh * TT + key0) * HD;
      const bf16_t* kp1 = kb + (bh * TT + key1) * HD;
#pragma unroll
      for (int c = 0; c < 2; ++c) {
        v16bf b0 = bfrag_contig(kp0 + c * 32 + grp * 16);
        v16bf b1 = bfrag_contig(kp1 + c * 32 + grp * 16);
        s0 = wmma_bf16(c ? qa1 : qa0, b0, s0);
        s1 = wmma_bf16(c ? qa1 : qa0, b1, s1);
      }
    }
    {  // stage V tile [32 keys][dims 0..31] -> LDS (one row per lane)
      const int key = min(k0 + (int)threadIdx.x, TT - 1);
      const bf16_t* vp = vb + (bh * TT + key) * HD;
#pragma unroll
      for (int j = 0; j < 4; ++j)
        *(v8bf*)&Vt[threadIdx.x * 72 + j * 8] = *(const v8bf*)(vp + j * 8);
    }
    float corr[8];
#pragma unroll
    for (int r = 0; r < 8; ++r) {
      const int qrow = qt * 16 + r + 8 * grp;
      float x0 = (k0 + ln) <= qrow ? s0[r] * 0.125f : -NEG_BIG;       // /sqrt(HD)
      float x1 = (k0 + 16 + ln) <= qrow ? s1[r] * 0.125f : -NEG_BIG;
      float mx = fmaxf(x0, x1);
#pragma unroll
      for (int off = 1; off < 16; off <<= 1) mx = fmaxf(mx, __shfl_xor(mx, off, 16));
      const float mn = fmaxf(mrow[r], mx);
      corr[r] = __expf(mrow[r] - mn);
      mrow[r] = mn;
      const float p0 = __expf(x0 - mn), p1 = __expf(x1 - mn);
      P[(r + 8 * grp) * 32 + ln] = (__bf16)p0;
      P[(r + 8 * grp) * 32 + 16 + ln] = (__bf16)p1;
      float rs = p0 + p1;
#pragma unroll
      for (int off = 1; off < 16; off <<= 1) rs += __shfl_xor(rs, off, 16);
      lrow[r] = lrow[r] * corr[r] + rs;
    }
#pragma unroll
    for (int r = 0; r < 8; ++r) {
      o0[r] *= corr[r]; o1[r] *= corr[r]; o2[r] *= corr[r]; o3[r] *= corr[r];
    }
    __syncthreads();
    const v16bf pa = cat8(*(const v8bf*)&P[ln * 32 + 16 * grp],
                          *(const v8bf*)&P[ln * 32 + 16 * grp + 8]);
#pragma unroll
    for (int ch = 0; ch < 4; ++ch) {
      v16bf vf;
#pragma unroll
      for (int e = 0; e < 16; ++e) vf[e] = Vt[(grp * 16 + e) * 72 + ch * 16 + ln];
      v8f& oc = ch == 0 ? o0 : ch == 1 ? o1 : ch == 2 ? o2 : o3;
      oc = wmma_bf16(pa, vf, oc);
    }
    __syncthreads();
  }
#pragma unroll
  for (int r = 0; r < 8; ++r) {
    const int trow = qt * 16 + r + 8 * grp;
    const float inv = 0.25f / lrow[r];  // /sqrt(H)
    const size_t base = ((size_t)b * TT + trow) * DD + h * HD;
    ob[base + 0 * 16 + ln] = (__bf16)(o0[r] * inv);
    ob[base + 1 * 16 + ln] = (__bf16)(o1[r] * inv);
    ob[base + 2 * 16 + ln] = (__bf16)(o2[r] * inv);
    ob[base + 3 * 16 + ln] = (__bf16)(o3[r] * inv);
  }
}

// ---------------- MoE gate logits (tiny N=8 GEMV per token) -----------------
__global__ void gate_kernel(const bf16_t* __restrict__ h2,
                            const float* __restrict__ gw,
                            const float* __restrict__ gb,
                            float* __restrict__ logits) {
  const int n = blockIdx.x * blockDim.x + threadIdx.x;
  if (n >= BT) return;
  float acc[EE] = {};
  const bf16_t* hr = h2 + (size_t)n * DD;
  for (int d = 0; d < DD; ++d) {
    const float hv = (float)hr[d];
    const float* g = gw + (size_t)d * EE;
#pragma unroll
    for (int e = 0; e < EE; ++e) acc[e] += hv * g[e];
  }
#pragma unroll
  for (int e = 0; e < EE; ++e) logits[(size_t)n * EE + e] = acc[e] + gb[e];
}

// ---------------- routing: top-2 + counts -----------------------------------
__global__ void route_topk(const float* __restrict__ logits, int* counts,
                           int* tok_e, float* tok_p) {
  const int n = blockIdx.x * blockDim.x + threadIdx.x;
  if (n >= BT) return;
  float l[EE];
#pragma unroll
  for (int e = 0; e < EE; ++e) l[e] = logits[(size_t)n * EE + e];
  int e1 = 0; float b1v = l[0];
#pragma unroll
  for (int e = 1; e < EE; ++e) if (l[e] > b1v) { b1v = l[e]; e1 = e; }
  int e2 = -1; float b2v = -NEG_BIG;
#pragma unroll
  for (int e = 0; e < EE; ++e)
    if (e != e1 && l[e] > b2v) { b2v = l[e]; e2 = e; }
  const float x2 = __expf(b2v - b1v);
  const float p1 = 1.0f / (1.0f + x2), p2 = x2 / (1.0f + x2);
  tok_e[n * 2] = e1; tok_e[n * 2 + 1] = e2;
  tok_p[n * 2] = p1; tok_p[n * 2 + 1] = p2;
  atomicAdd(&counts[e1], 1);
  atomicAdd(&counts[e2], 1);
}

__global__ void route_scan(const int* __restrict__ counts, int* offs,
                           int* row_expert) {
  if (threadIdx.x || blockIdx.x) return;
  int run = 0;
  for (int e = 0; e < EE; ++e) {
    offs[e] = run;
    const int padded = ((counts[e] + 15) >> 4) << 4;  // 16-align expert groups
    for (int r = 0; r < padded; ++r) row_expert[run + r] = e;
    run += padded;
  }
  for (int r = run; r < PAIR_CAP; ++r) row_expert[r] = EE - 1;
}

__global__ void route_assign(const int* __restrict__ tok_e,
                             const float* __restrict__ tok_p,
                             const int* __restrict__ offs, int* fill,
                             int* pair_tok, float* pair_wt) {
  const int n = blockIdx.x * blockDim.x + threadIdx.x;
  if (n >= BT) return;
#pragma unroll
  for (int k = 0; k < KTOP; ++k) {
    const int e = tok_e[n * 2 + k];
    const int slot = offs[e] + atomicAdd(&fill[e], 1);
    pair_tok[slot] = n;
    pair_wt[slot] = tok_p[n * 2 + k];
  }
}

// ---------------- MoE FFN stage 1: a = silu(h2@w1+b1) * (h2@w2+b2) ----------
// wave tile: 16 gathered rows x 16 FF cols, dual accumulators, K over D.
__global__ void moe_ffn1(const bf16_t* __restrict__ h2,
                         const float* __restrict__ w1, const float* __restrict__ b1,
                         const float* __restrict__ w2, const float* __restrict__ b2,
                         const int* __restrict__ pair_tok,
                         const int* __restrict__ row_expert,
                         bf16_t* __restrict__ abuf) {
  const int tid = threadIdx.x, lane = tid & 31, wid = tid >> 5;
  const int grp = lane >> 4, ln = lane & 15;
  const int r0 = blockIdx.y * 16;
  const int ffb = (blockIdx.x * 8 + wid) * 16;
  const int tok = pair_tok[r0 + ln];
  if (__all(tok < 0)) return;  // unused padded tile
  const int e = row_expert[r0];
  const bf16_t* Arow = h2 + (size_t)(tok < 0 ? 0 : tok) * DD;
  const float* W1 = w1 + (size_t)e * DD * FF;
  const float* W2 = w2 + (size_t)e * DD * FF;
  v8f a1 = {}, a2 = {};
  for (int k0 = 0; k0 < DD; k0 += 32) {
    const v16bf a = afrag(Arow, k0, grp);
    const float* w1p = W1 + (size_t)(k0 + grp * 16) * FF + ffb + ln;
    const float* w2p = W2 + (size_t)(k0 + grp * 16) * FF + ffb + ln;
    if (k0 + 32 < DD) {
      __builtin_prefetch(w1p + (size_t)32 * FF, 0, 1);
      __builtin_prefetch(w2p + (size_t)32 * FF, 0, 1);
    }
    a1 = wmma_bf16(a, bfrag_strided_f32(w1p, FF), a1);
    a2 = wmma_bf16(a, bfrag_strided_f32(w2p, FF), a2);
  }
  const int col = ffb + ln;
#pragma unroll
  for (int r = 0; r < 8; ++r) {
    const int row = r0 + r + 8 * grp;
    if (pair_tok[row] >= 0) {
      const float x1v = a1[r] + b1[(size_t)e * FF + col];
      const float x2v = a2[r] + b2[(size_t)e * FF + col];
      const float sil = x1v / (1.0f + __expf(-x1v));
      abuf[(size_t)row * FF + col] = (__bf16)(sil * x2v);
    }
  }
}

// ---------------- MoE FFN stage 2: moe[tok] += wt * (a@wp + bp) -------------
__global__ void moe_ffn2(const bf16_t* __restrict__ abuf,
                         const float* __restrict__ wp, const float* __restrict__ bp,
                         const int* __restrict__ pair_tok,
                         const float* __restrict__ pair_wt,
                         const int* __restrict__ row_expert,
                         float* __restrict__ moe) {
  const int tid = threadIdx.x, lane = tid & 31, wid = tid >> 5;
  const int grp = lane >> 4, ln = lane & 15;
  const int r0 = blockIdx.y * 16;
  const int nb = (blockIdx.x * 8 + wid) * 16;
  const int tok = pair_tok[r0 + ln];
  if (__all(tok < 0)) return;
  const int e = row_expert[r0];
  const bf16_t* Arow = abuf + (size_t)(r0 + ln) * FF;
  const float* Wp = wp + (size_t)e * FF * DD;
  v8f acc = {};
  for (int k0 = 0; k0 < FF; k0 += 32) {
    const v16bf a = afrag(Arow, k0, grp);
    const float* wpp = Wp + (size_t)(k0 + grp * 16) * DD + nb + ln;
    if (k0 + 32 < FF) __builtin_prefetch(wpp + (size_t)32 * DD, 0, 1);
    acc = wmma_bf16(a, bfrag_strided_f32(wpp, DD), acc);
  }
  const int col = nb + ln;
#pragma unroll
  for (int r = 0; r < 8; ++r) {
    const int row = r0 + r + 8 * grp;
    const int tr = pair_tok[row];
    if (tr >= 0) {
      const float val = (acc[r] + bp[(size_t)e * DD + col]) * pair_wt[row];
      atomic_add_f32(&moe[(size_t)tr * DD + col], val);
    }
  }
}

// ---------------- final: out = x2 + moe; out[BT*D] = aux_loss ---------------
__global__ void final_add(const float* __restrict__ x2,
                          const float* __restrict__ moe,
                          const int* __restrict__ counts, float* __restrict__ out) {
  const size_t i = (size_t)blockIdx.x * blockDim.x + threadIdx.x;
  if (i < (size_t)BT * DD) out[i] = x2[i] + moe[i];
  if (i == 0) {
    float aux = 0.0f;
    for (int e = 0; e < EE; ++e) {
      const float f = (float)counts[e] / (float)(BT * KTOP) - 1.0f / (float)EE;
      aux += f * f;
    }
    out[(size_t)BT * DD] = aux;
  }
}

// ---------------- launcher ---------------------------------------------------
extern "C" void kernel_launch(void* const* d_in, const int* in_sizes, int n_in,
                              void* d_out, int out_size, void* d_ws, size_t ws_size,
                              hipStream_t stream) {
  const float* x        = (const float*)d_in[0];
  const float* n1scale  = (const float*)d_in[1];
  const float* n2scale  = (const float*)d_in[2];
  const float* qkv_w    = (const float*)d_in[3];
  const float* qkv_b    = (const float*)d_in[4];
  const float* out_w    = (const float*)d_in[5];
  const float* out_b    = (const float*)d_in[6];
  const float* gate_w   = (const float*)d_in[7];
  const float* gate_b   = (const float*)d_in[8];
  const float* w1       = (const float*)d_in[9];
  const float* b1       = (const float*)d_in[10];
  const float* w2       = (const float*)d_in[11];
  const float* b2       = (const float*)d_in[12];
  const float* wp       = (const float*)d_in[13];
  const float* bp       = (const float*)d_in[14];
  float* out = (float*)d_out;
  char* ws = (char*)d_ws;

  bf16_t* hbuf   = (bf16_t*)(ws + O_H);
  float*  qkvbuf = (float*)(ws + O_QKV);
  bf16_t* obuf   = (bf16_t*)(ws + O_O);
  bf16_t* qbuf   = (bf16_t*)(ws + O_Q);
  bf16_t* kbuf   = (bf16_t*)(ws + O_K);
  bf16_t* vbuf   = (bf16_t*)(ws + O_V);
  float*  x2buf  = (float*)(ws + O_X2);
  bf16_t* h2buf  = (bf16_t*)(ws + O_H2);
  char*   rt     = ws + O_ROUTE;
  int*    counts = (int*)(rt + R_COUNTS);
  int*    offs   = (int*)(rt + R_OFFS);
  int*    fill   = (int*)(rt + R_FILL);
  int*    tok_e  = (int*)(rt + R_TOKE);
  float*  tok_p  = (float*)(rt + R_TOKP);
  int*    ptok   = (int*)(rt + R_PTOK);
  float*  pwt    = (float*)(rt + R_PWT);
  int*    rexp   = (int*)(rt + R_REXP);
  float*  moebuf = (float*)(ws + O_MOE);
  bf16_t* abuf   = (bf16_t*)(ws + O_ABUF);  // aliases dead h/qkv/o/q regions

  init_kernel<<<4096, 256, 0, stream>>>(moebuf, ptok, fill, counts);
  rmsnorm_kernel<<<BT, 256, 0, stream>>>(x, n1scale, hbuf);
  gemm_bf16<<<dim3(N3 / 32, BT / 64), 256, 0, stream>>>(hbuf, qkv_w, qkv_b,
                                                        nullptr, qkvbuf, N3, DD);
  rope_split<<<(BT * HH * 32) / 256, 256, 0, stream>>>(qkvbuf, qbuf, kbuf, vbuf);
  attn_kernel<<<dim3(TT / 16, HH, BB), 32, 0, stream>>>(qbuf, kbuf, vbuf, obuf);
  gemm_bf16<<<dim3(DD / 32, BT / 64), 256, 0, stream>>>(obuf, out_w, out_b, x,
                                                        x2buf, DD, DD);
  rmsnorm_kernel<<<BT, 256, 0, stream>>>(x2buf, n2scale, h2buf);
  // logits go to qkvbuf (dead after rope_split; consumed by route_topk before
  // moe_ffn1 overwrites that region via the abuf alias)
  gate_kernel<<<BT / 256, 256, 0, stream>>>(h2buf, gate_w, gate_b, qkvbuf);
  route_topk<<<BT / 256, 256, 0, stream>>>(qkvbuf, counts, tok_e, tok_p);
  route_scan<<<1, 1, 0, stream>>>(counts, offs, rexp);
  route_assign<<<BT / 256, 256, 0, stream>>>(tok_e, tok_p, offs, fill, ptok, pwt);
  moe_ffn1<<<dim3(FF / 128, PAIR_TILES), 256, 0, stream>>>(h2buf, w1, b1, w2, b2,
                                                           ptok, rexp, abuf);
  moe_ffn2<<<dim3(DD / 128, PAIR_TILES), 256, 0, stream>>>(abuf, wp, bp, ptok,
                                                           pwt, rexp, moebuf);
  final_add<<<(BT * DD) / 256, 256, 0, stream>>>(x2buf, moebuf, counts, out);
}
